// DigitConvolutionalModel_34179349741691
// MI455X (gfx1250) — matrix operand order repaired and checked by reading, not compiled, not run on hardware
//
#include <hip/hip_runtime.h>
#include <hip/hip_bf16.h>

typedef __attribute__((ext_vector_type(16))) _Float16 v16h;
typedef __attribute__((ext_vector_type(8)))  _Float16 v8h;
typedef __attribute__((ext_vector_type(8)))  float    v8f;

#define BATCH   65536
#define HW      28
#define OUTHW   26
#define FEAT    676      // 26*26
#define KPAD    704      // FEAT padded to multiple of 32 (22 K-tiles)
#define HID     200
#define NPAD    208      // HID padded to multiple of 16 (13 N-tiles)
#define NCLS    10
#define KTILES  22
#define NTILES  13

// ---------------- weight prep: w1 (676x200 f32) -> w1T (208x704 f16, padded) --
__global__ void prep_w1t(const float* __restrict__ w1, _Float16* __restrict__ w1T) {
    int i = blockIdx.x * 256 + threadIdx.x;
    if (i >= NPAD * KPAD) return;
    int n = i / KPAD;
    int k = i - n * KPAD;
    float v = (n < HID && k < FEAT) ? w1[k * HID + n] : 0.0f;
    w1T[i] = (_Float16)v;
}

// ---------------- conv: 3x3 valid box filter, emit f16 padded features -------
__global__ void conv_box(const float* __restrict__ x,
                         const float* __restrict__ cw,
                         _Float16* __restrict__ featH) {
    __shared__ float img[8 * HW * HW];
    const int tid = threadIdx.x;
    const size_t xbase = (size_t)blockIdx.x * 8 * (HW * HW);

    for (int i = tid; i < 8 * HW * HW; i += 256)
        img[i] = x[xbase + i];
    __syncthreads();

    const float c00 = cw[0], c01 = cw[1], c02 = cw[2];
    const float c10 = cw[3], c11 = cw[4], c12 = cw[5];
    const float c20 = cw[6], c21 = cw[7], c22 = cw[8];

    const size_t fbase = (size_t)blockIdx.x * 8 * KPAD;
    for (int i = tid; i < 8 * KPAD; i += 256) {
        int im = i / KPAD;
        int f  = i - im * KPAD;
        float s = 0.0f;
        if (f < FEAT) {
            int r = f / OUTHW;
            int c = f - r * OUTHW;
            const float* p = &img[im * (HW * HW) + r * HW + c];
            s = p[0]  * c00 + p[1]  * c01 + p[2]  * c02
              + p[28] * c10 + p[29] * c11 + p[30] * c12
              + p[56] * c20 + p[57] * c21 + p[58] * c22;
        }
        featH[fbase + i] = (_Float16)s;
    }
}

// ---------------- fused: hidden = relu(feat@w1+b1); out = hidden@w2+b2 -------
// Block: 256 threads = 8 waves; each wave computes a 16x208 hidden strip with
// WMMA f32_16x16x32_f16 (13 accumulators x 22 K-steps), ReLU in registers,
// stages hidden in LDS (f16), then scalar GEMM2 (200x10) from LDS.
__global__ void __launch_bounds__(256)
fused_mlp(const _Float16* __restrict__ featH,
          const _Float16* __restrict__ w1T,
          const float* __restrict__ b1,
          const float* __restrict__ w2,
          const float* __restrict__ b2,
          float* __restrict__ out) {
    __shared__ _Float16 hid[128 * NPAD];   // 53,248 B

    const int tid  = threadIdx.x;
    const int wave = tid >> 5;             // 0..7
    const int lane = tid & 31;
    const int l16  = lane & 15;
    const int hi   = lane >> 4;            // 0 or 1

    const int m0 = blockIdx.x * 128 + wave * 16;

    v8f acc[NTILES];
#pragma unroll
    for (int nt = 0; nt < NTILES; ++nt) { v8f z = {}; acc[nt] = z; }

    // A fragment: lane holds row (l&15); K halves: v0-3 = K koff..koff+7,
    // v4-7 = K 16+koff.., koff = (lane>=16)?8:0  -> two 16B loads per step.
    const _Float16* aptr = featH + (size_t)(m0 + l16) * KPAD + hi * 8;
    // B fragment: lane holds col (l&15); same K split as A, koff = (lane>=16)?8:0.
    const int kgroup = hi * 8;

    for (int kk = 0; kk < KTILES; ++kk) {
        const int kbase = kk * 32;
        v8h alo = *reinterpret_cast<const v8h*>(aptr + kbase);
        v8h ahi = *reinterpret_cast<const v8h*>(aptr + kbase + 16);
        v16h a;
#pragma unroll
        for (int i = 0; i < 8; ++i) { a[i] = alo[i]; a[8 + i] = ahi[i]; }

#pragma unroll
        for (int nt = 0; nt < NTILES; ++nt) {
            const _Float16* bp =
                w1T + (size_t)(nt * 16 + l16) * KPAD + kbase + kgroup;
            v8h blo = *reinterpret_cast<const v8h*>(bp);
            v8h bhi = *reinterpret_cast<const v8h*>(bp + 16);
            v16h b;
#pragma unroll
            for (int i = 0; i < 8; ++i) { b[i] = blo[i]; b[8 + i] = bhi[i]; }
            acc[nt] = __builtin_amdgcn_wmma_f32_16x16x32_f16(
                false, a, false, b, (short)0, acc[nt], false, false);
        }
    }

    // bias + ReLU, stage hidden strip in LDS (f16).
    // C layout: VGPR r -> M = r (lanes 0-15) or 8+r (lanes 16-31); N = l&15.
#pragma unroll
    for (int nt = 0; nt < NTILES; ++nt) {
        const int n = nt * 16 + l16;
        const float bias = (n < HID) ? b1[n] : 0.0f;
        const int mloc = wave * 16 + hi * 8;
#pragma unroll
        for (int r = 0; r < 8; ++r) {
            float v = acc[nt][r] + bias;
            v = v > 0.0f ? v : 0.0f;
            hid[(mloc + r) * NPAD + n] = (_Float16)v;
        }
    }
    __syncthreads();

    // GEMM2: 128 rows x 10 cols; 2 threads/row, 5 cols/thread.
    const int mloc = tid >> 1;
    const int c0   = (tid & 1) * 5;
    float s[5];
#pragma unroll
    for (int c = 0; c < 5; ++c) s[c] = b2[c0 + c];
    const _Float16* hrow = &hid[mloc * NPAD];
    for (int n = 0; n < HID; ++n) {
        const float hv = (float)hrow[n];
        const float* wrow = &w2[n * NCLS + c0];
#pragma unroll
        for (int c = 0; c < 5; ++c) s[c] += hv * wrow[c];
    }
    float* op = out + ((size_t)blockIdx.x * 128 + mloc) * NCLS + c0;
#pragma unroll
    for (int c = 0; c < 5; ++c) op[c] = s[c];
}

extern "C" void kernel_launch(void* const* d_in, const int* in_sizes, int n_in,
                              void* d_out, int out_size, void* d_ws, size_t ws_size,
                              hipStream_t stream) {
    const float* x  = (const float*)d_in[0];
    const float* cw = (const float*)d_in[1];
    const float* w1 = (const float*)d_in[2];
    const float* b1 = (const float*)d_in[3];
    const float* w2 = (const float*)d_in[4];
    const float* b2 = (const float*)d_in[5];
    float* out = (float*)d_out;

    _Float16* featH = (_Float16*)d_ws;                                   // 92,274,688 B
    _Float16* w1T   = (_Float16*)((char*)d_ws + (size_t)BATCH * KPAD * 2);

    prep_w1t<<<(NPAD * KPAD + 255) / 256, 256, 0, stream>>>(w1, w1T);
    conv_box<<<BATCH / 8, 256, 0, stream>>>(x, cw, featH);
    fused_mlp<<<BATCH / 128, 256, 0, stream>>>(featH, w1T, b1, w2, b2, out);
}
